// AstroEconomicTransformer_2714419331447
// MI455X (gfx1250) — compile-verified, matrix operand order checked by reading, not run on hardware
//
#include <hip/hip_runtime.h>
#include <hip/hip_bf16.h>

// ---------------------------------------------------------------------------
// Model constants (match reference)
// ---------------------------------------------------------------------------
#define BB   2
#define SS   1024
#define DD   1024
#define HH   16
#define LL   6
#define DFF  4096
#define NM   10
#define NA   20
#define HDIM 64
#define EPSF 1e-5f

typedef __attribute__((ext_vector_type(16))) __bf16          v16bf;
typedef __attribute__((ext_vector_type(8)))  float           v8f;
typedef __attribute__((ext_vector_type(16))) unsigned short  v16us;
typedef __attribute__((ext_vector_type(4)))  unsigned int    v4u;
typedef __attribute__((ext_vector_type(8)))  int             v8i;
typedef __attribute__((ext_vector_type(4)))  int             v4i;

// ---------------------------------------------------------------------------
// Helpers
// ---------------------------------------------------------------------------
__device__ __forceinline__ unsigned short f2bf(float x) {
  unsigned int u = __builtin_bit_cast(unsigned int, x);
  u += 0x7FFFu + ((u >> 16) & 1u);           // round-to-nearest-even
  return (unsigned short)(u >> 16);
}

// Load one 16-bit WMMA fragment (A or B operand) per ISA 7.12.2 striping:
// lanes 0-15 carry rows 0-15; lane-half selects K-subgroup.
// Per lane: halves [0..7] = p[klane .. klane+7], halves [8..15] = p[klane+16 ..].
__device__ __forceinline__ v16bf fragK(const unsigned short* p, int klane) {
  v16us u;
#pragma unroll
  for (int i = 0; i < 8; ++i) {
    u[i]     = p[klane + i];
    u[i + 8] = p[klane + 16 + i];
  }
  return __builtin_bit_cast(v16bf, u);
}

__device__ __forceinline__ v8f wmma_bf16(v16bf a, v16bf b, v8f c) {
  return __builtin_amdgcn_wmma_f32_16x16x32_bf16(
      /*neg_a=*/false, a, /*neg_b=*/false, b,
      /*c_mod=*/(short)0, c, /*reuse_a=*/false, /*reuse_b=*/false);
}

// ---------------------------------------------------------------------------
// Tensor Data Mover: async DMA of a 2-D bf16 tile (tile_d0 x tile_d1 elems,
// row stride stride0 elems) from global memory into contiguous LDS.
// D# layout per CDNA5 ISA ch.8: group0 = count/lds_addr/global_addr/type,
// group1 = data_size, tensor dims, tile dims, dim0 stride. Groups 2-3 zero
// (2-D tensor: tile_dim2..4 = 0 -> unused). Tracked by TENSORcnt.
// This toolchain exposes the 6-arg builtin (extra int32x8 group: zeroed).
// ---------------------------------------------------------------------------
__device__ __forceinline__ void tdm_load_2d(const void* gaddr,
                                            unsigned lds_addr,
                                            unsigned tile_d0, unsigned tile_d1,
                                            unsigned long long tensor_d0,
                                            unsigned long long tensor_d1,
                                            unsigned long long stride0) {
  unsigned long long ga = (unsigned long long)(size_t)gaddr;
  v4u g0;
  g0[0] = 1u;                                    // count=1, is_restore=0, no gather
  g0[1] = lds_addr;                              // LDS byte address
  g0[2] = (unsigned)ga;                          // global_addr[31:0]
  g0[3] = (unsigned)((ga >> 32) & 0x01FFFFFFu)   // global_addr[56:32]
          | (2u << 30);                          // type = 2 ("image")
  v8i g1;
  g1[0] = (int)(1u << 16);                       // data_size = 1 -> 2 bytes
  g1[1] = (int)((tensor_d0 & 0xFFFFu) << 16);    // tensor_dim0[15:0] @ bit48
  g1[2] = (int)(((tensor_d0 >> 16) & 0xFFFFu) |  // tensor_dim0[31:16]
                ((tensor_d1 & 0xFFFFu) << 16));  // tensor_dim1[15:0]
  g1[3] = (int)(((tensor_d1 >> 16) & 0xFFFFu) |  // tensor_dim1[31:16]
                (tile_d0 << 16));                // tile_dim0
  g1[4] = (int)(tile_d1 & 0xFFFFu);              // tile_dim1 (tile_dim2 = 0)
  g1[5] = (int)(stride0 & 0xFFFFFFFFu);          // tensor_dim0_stride[31:0]
  g1[6] = (int)((stride0 >> 32) & 0xFFFFu);      // stride[47:32]; dim1_stride=0
  g1[7] = 0;
  v4i z4 = {0, 0, 0, 0};
  v8i z8 = {0, 0, 0, 0, 0, 0, 0, 0};
  __builtin_amdgcn_tensor_load_to_lds(g0, g1, z4, z4, z8, 0);
}

// ---------------------------------------------------------------------------
// fp32 -> bf16 conversion
// ---------------------------------------------------------------------------
__global__ void cvt_bf16_kernel(const float* __restrict__ in,
                                unsigned short* __restrict__ out, int n) {
  int i = blockIdx.x * blockDim.x + threadIdx.x;
  if (i < n) out[i] = f2bf(in[i]);
}

// ---------------------------------------------------------------------------
// Embedding: x = concat(mkt@Wm.T+bm, ast@Wa.T+ba) + pe   -> fp32 + bf16
// ---------------------------------------------------------------------------
__global__ void embed_kernel(const float* __restrict__ mkt,
                             const float* __restrict__ ast,
                             const float* __restrict__ pe,
                             const float* __restrict__ Wm, const float* __restrict__ bm,
                             const float* __restrict__ Wa, const float* __restrict__ ba,
                             float* __restrict__ x32, unsigned short* __restrict__ xb) {
  int idx = blockIdx.x * blockDim.x + threadIdx.x;
  if (idx >= BB * SS * DD) return;
  int d  = idx % DD;
  int bs = idx / DD;
  int s  = bs % SS;
  float acc;
  if (d < DD / 2) {
    acc = bm[d];
    const float* w = Wm + (size_t)d * NM;
    const float* m = mkt + (size_t)bs * NM;
#pragma unroll
    for (int j = 0; j < NM; ++j) acc += m[j] * w[j];
  } else {
    int dd = d - DD / 2;
    acc = ba[dd];
    const float* w = Wa + (size_t)dd * NA;
    const float* a = ast + (size_t)bs * NA;
#pragma unroll
    for (int j = 0; j < NA; ++j) acc += a[j] * w[j];
  }
  acc += pe[(size_t)s * DD + d];
  x32[idx] = acc;
  xb[idx]  = f2bf(acc);
}

// ---------------------------------------------------------------------------
// WMMA GEMM:  C[M,N] = A[M,K](bf16) @ W[N,K](bf16)^T  + bias (+resid) (+relu)
// Block tile 128x128, 8 waves (2x4), wave tile 64x32, K-step 32.
// LDS staging via TDM (tensor_load_to_lds), double-buffered: wave 0 issues
// the next tile's DMA, waits tensorcnt<=2 (in-order => current tile done),
// and the workgroup barrier publishes LDS to the other waves.
// ---------------------------------------------------------------------------
#define GTM 128
#define GTN 128
#define GTK 32

__global__ __launch_bounds__(256) void gemm_bf16_kernel(
    const unsigned short* __restrict__ A,  // M x K
    const unsigned short* __restrict__ W,  // N x K
    const float* __restrict__ bias,        // N (may be null)
    const float* __restrict__ resid,       // M x N (may be null)
    float* __restrict__ outF,              // M x N (may be null)
    unsigned short* __restrict__ outH,     // M x N (may be null)
    int M, int N, int K, int relu) {
  __shared__ unsigned short As[2][GTM][GTK];
  __shared__ unsigned short Ws[2][GTN][GTK];

  const int tid  = threadIdx.x;
  const int wave = tid >> 5;
  const int lane = tid & 31;
  const int lm   = lane & 15;
  const int klane = ((lane >> 4) & 1) * 8;
  const int wm = wave >> 2;   // 0..1  (M direction, 64 rows each)
  const int wn = wave & 3;    // 0..3  (N direction, 32 cols each)
  const int mblk = blockIdx.x * GTM;
  const int nblk = blockIdx.y * GTN;
  const int nk   = K / GTK;

  v8f acc[4][2] = {};

  if (wave == 0) {   // uniform branch: only wave 0 drives the TDM
    tdm_load_2d(A + (size_t)mblk * K, (unsigned)(size_t)&As[0][0][0],
                GTK, GTM, (unsigned long long)K, (unsigned long long)M,
                (unsigned long long)K);
    tdm_load_2d(W + (size_t)nblk * K, (unsigned)(size_t)&Ws[0][0][0],
                GTK, GTN, (unsigned long long)K, (unsigned long long)N,
                (unsigned long long)K);
  }

  for (int k = 0; k < nk; ++k) {
    const int cur = k & 1;
    if (wave == 0) {
      if (k + 1 < nk) {
        tdm_load_2d(A + (size_t)mblk * K + (size_t)(k + 1) * GTK,
                    (unsigned)(size_t)&As[cur ^ 1][0][0],
                    GTK, GTM, (unsigned long long)K, (unsigned long long)M,
                    (unsigned long long)K);
        tdm_load_2d(W + (size_t)nblk * K + (size_t)(k + 1) * GTK,
                    (unsigned)(size_t)&Ws[cur ^ 1][0][0],
                    GTK, GTN, (unsigned long long)K, (unsigned long long)N,
                    (unsigned long long)K);
        __builtin_amdgcn_s_wait_tensorcnt(2);  // current tile complete
      } else {
        __builtin_amdgcn_s_wait_tensorcnt(0);
      }
    }
    __syncthreads();                            // publish buf[cur]

    v16bf bfrag[2];
#pragma unroll
    for (int nt = 0; nt < 2; ++nt)
      bfrag[nt] = fragK(&Ws[cur][wn * 32 + nt * 16 + lm][0], klane);
#pragma unroll
    for (int mt = 0; mt < 4; ++mt) {
      v16bf afrag = fragK(&As[cur][wm * 64 + mt * 16 + lm][0], klane);
#pragma unroll
      for (int nt = 0; nt < 2; ++nt)
        acc[mt][nt] = wmma_bf16(afrag, bfrag[nt], acc[mt][nt]);
    }
    __syncthreads();   // all waves done with buf[cur] before it is re-DMA'd
  }

  // Epilogue: C layout = lane 0-15 -> col, vgpr r -> row r (+8 for hi half).
  const int lh = (lane >> 4) & 1;
#pragma unroll
  for (int mt = 0; mt < 4; ++mt) {
#pragma unroll
    for (int nt = 0; nt < 2; ++nt) {
      int col = nblk + wn * 32 + nt * 16 + lm;
      float bv = bias ? bias[col] : 0.0f;
#pragma unroll
      for (int r = 0; r < 8; ++r) {
        int row = mblk + wm * 64 + mt * 16 + r + lh * 8;
        float v = acc[mt][nt][r] + bv;
        if (resid) v += resid[(size_t)row * N + col];
        if (relu)  v = v > 0.0f ? v : 0.0f;
        if (outF) outF[(size_t)row * N + col] = v;
        if (outH) outH[(size_t)row * N + col] = f2bf(v);
      }
    }
  }
}

// ---------------------------------------------------------------------------
// Fused flash-style attention (per (b, h, 64-query block); 4 waves).
// scores = QK^T / 8 (per-head scalar bias cancels in softmax); online softmax;
// ctx = P V. K block staged by TDM; V transposed manually (TDM can't).
// ---------------------------------------------------------------------------
__global__ __launch_bounds__(128) void attn_kernel(
    const unsigned short* __restrict__ Q,   // (B*S, D) bf16
    const unsigned short* __restrict__ Km,  // (B*S, D) bf16
    const unsigned short* __restrict__ Vm,  // (B*S, D) bf16
    unsigned short* __restrict__ ctx) {     // (B*S, D) bf16
  __shared__ unsigned short Ks[64][HDIM];        // K block, row-major
  __shared__ unsigned short Vt[HDIM][64];        // V block transposed [d][key]
  __shared__ unsigned short Ps[4][16][64];       // per-wave P staging (transpose)

  const int qblk = blockIdx.x;          // 0..15
  const int h    = blockIdx.y;          // 0..15
  const int b    = blockIdx.z;          // 0..1
  const int tid  = threadIdx.x;
  const int wave = tid >> 5;
  const int lane = tid & 31;
  const int lm   = lane & 15;
  const int lh   = (lane >> 4) & 1;
  const int klane = lh * 8;

  const size_t rowbase = (size_t)b * SS;
  const int qrow0 = qblk * 64 + wave * 16;

  // Q A-fragments held for the whole kernel (HD=64 -> two K=32 chunks).
  v16bf aq[2];
  {
    const unsigned short* qp =
        Q + (rowbase + qrow0 + lm) * (size_t)DD + h * HDIM;
    aq[0] = fragK(qp, klane);
    aq[1] = fragK(qp + 32, klane);
  }

  float rmax[8], rsum[8];
#pragma unroll
  for (int r = 0; r < 8; ++r) { rmax[r] = -1e30f; rsum[r] = 0.0f; }
  v8f occ[4] = {};
  const float scale = 0.125f;           // 1/sqrt(64)

  for (int kb = 0; kb < SS / 64; ++kb) {
    // Stage K block via TDM (64x64 bf16 tile, row stride D elems).
    if (wave == 0) {
      tdm_load_2d(Km + (rowbase + kb * 64) * (size_t)DD + h * HDIM,
                  (unsigned)(size_t)&Ks[0][0], HDIM, 64,
                  (unsigned long long)DD, (unsigned long long)(BB * SS),
                  (unsigned long long)DD);
    }
    // Stage V block transposed into LDS (manual: DMA cannot transpose).
    for (int i = tid; i < 64 * HDIM; i += 128) {
      int kr = i >> 6, kc = i & 63;
      Vt[kc][kr] = Vm[(rowbase + kb * 64 + kr) * (size_t)DD + h * HDIM + kc];
    }
    if (wave == 0) __builtin_amdgcn_s_wait_tensorcnt(0);
    __syncthreads();

    // S tile: 16 queries x 64 keys (4 WMMA N-subtiles, 2 K-steps each).
    v8f sc[4] = {};
#pragma unroll
    for (int nt = 0; nt < 4; ++nt) {
      const unsigned short* kp = &Ks[nt * 16 + lm][0];
      sc[nt] = wmma_bf16(aq[0], fragK(kp, klane), sc[nt]);
      sc[nt] = wmma_bf16(aq[1], fragK(kp + 32, klane), sc[nt]);
    }

    // Online softmax. Each lane-half tracks 8 rows (m = r + lh*8).
    float alpha[8], psum[8];
#pragma unroll
    for (int r = 0; r < 8; ++r) {
      float m = -1e30f;
#pragma unroll
      for (int nt = 0; nt < 4; ++nt) m = fmaxf(m, sc[nt][r] * scale);
#pragma unroll
      for (int msk = 1; msk < 16; msk <<= 1) m = fmaxf(m, __shfl_xor(m, msk, 32));
      float mn = fmaxf(rmax[r], m);
      alpha[r] = __expf(rmax[r] - mn);
      rmax[r]  = mn;
      psum[r]  = 0.0f;
    }
#pragma unroll
    for (int nt = 0; nt < 4; ++nt) {
#pragma unroll
      for (int r = 0; r < 8; ++r) {
        float p = __expf(sc[nt][r] * scale - rmax[r]);
        psum[r] += p;
        Ps[wave][r + lh * 8][nt * 16 + lm] = f2bf(p);
      }
    }
#pragma unroll
    for (int r = 0; r < 8; ++r) {
#pragma unroll
      for (int msk = 1; msk < 16; msk <<= 1) psum[r] += __shfl_xor(psum[r], msk, 32);
      rsum[r] = rsum[r] * alpha[r] + psum[r];
    }
#pragma unroll
    for (int dt = 0; dt < 4; ++dt)
#pragma unroll
      for (int r = 0; r < 8; ++r) occ[dt][r] *= alpha[r];

    // ctx += P @ V   (A from wave-local Ps, B from Vt[d][key]).
#pragma unroll
    for (int kk = 0; kk < 2; ++kk) {
      v16bf ap = fragK(&Ps[wave][lm][kk * 32], klane);
#pragma unroll
      for (int dt = 0; dt < 4; ++dt) {
        v16bf bv = fragK(&Vt[dt * 16 + lm][kk * 32], klane);
        occ[dt] = wmma_bf16(ap, bv, occ[dt]);
      }
    }
    __syncthreads();
  }

  // Normalize and write ctx (bf16) for the output projection GEMM.
#pragma unroll
  for (int dt = 0; dt < 4; ++dt) {
#pragma unroll
    for (int r = 0; r < 8; ++r) {
      size_t row = rowbase + qrow0 + r + lh * 8;
      int col = h * HDIM + dt * 16 + lm;
      ctx[row * (size_t)DD + col] = f2bf(occ[dt][r] / rsum[r]);
    }
  }
}

// ---------------------------------------------------------------------------
// LayerNorm over last dim (D=1024). One 256-thread block per row.
// Writes fp32 (for residual) and bf16 (for next GEMM A), either optional.
// ---------------------------------------------------------------------------
__global__ __launch_bounds__(256) void ln_kernel(
    const float* __restrict__ in, const float* __restrict__ g,
    const float* __restrict__ bta, float* __restrict__ outF,
    unsigned short* __restrict__ outH) {
  __shared__ float red[256];
  const int row = blockIdx.x;
  const float* x = in + (size_t)row * DD;

  float s = 0.0f;
  for (int i = threadIdx.x; i < DD; i += 256) s += x[i];
  red[threadIdx.x] = s;
  __syncthreads();
  for (int off = 128; off > 0; off >>= 1) {
    if (threadIdx.x < off) red[threadIdx.x] += red[threadIdx.x + off];
    __syncthreads();
  }
  float mean = red[0] * (1.0f / DD);
  __syncthreads();

  float s2 = 0.0f;
  for (int i = threadIdx.x; i < DD; i += 256) {
    float d = x[i] - mean;
    s2 += d * d;
  }
  red[threadIdx.x] = s2;
  __syncthreads();
  for (int off = 128; off > 0; off >>= 1) {
    if (threadIdx.x < off) red[threadIdx.x] += red[threadIdx.x + off];
    __syncthreads();
  }
  float rstd = rsqrtf(red[0] * (1.0f / DD) + EPSF);

  for (int i = threadIdx.x; i < DD; i += 256) {
    float v = (x[i] - mean) * rstd * g[i] + bta[i];
    if (outF) outF[(size_t)row * DD + i] = v;
    if (outH) outH[(size_t)row * DD + i] = f2bf(v);
  }
}

// ---------------------------------------------------------------------------
// Final projection D -> 1. One block per row.
// ---------------------------------------------------------------------------
__global__ __launch_bounds__(256) void out_kernel(
    const float* __restrict__ x, const float* __restrict__ Wout,
    const float* __restrict__ bout, float* __restrict__ out) {
  __shared__ float red[256];
  const int row = blockIdx.x;
  const float* xr = x + (size_t)row * DD;
  float s = 0.0f;
  for (int i = threadIdx.x; i < DD; i += 256) s += xr[i] * Wout[i];
  red[threadIdx.x] = s;
  __syncthreads();
  for (int off = 128; off > 0; off >>= 1) {
    if (threadIdx.x < off) red[threadIdx.x] += red[threadIdx.x + off];
    __syncthreads();
  }
  if (threadIdx.x == 0) out[row] = red[0] + bout[0];
}

// ---------------------------------------------------------------------------
// Host orchestration
// ---------------------------------------------------------------------------
static inline void cvt(const float* src, unsigned short* dst, int n,
                       hipStream_t stream) {
  cvt_bf16_kernel<<<(n + 255) / 256, 256, 0, stream>>>(src, dst, n);
}

extern "C" void kernel_launch(void* const* d_in, const int* in_sizes, int n_in,
                              void* d_out, int out_size, void* d_ws,
                              size_t ws_size, hipStream_t stream) {
  (void)in_sizes; (void)n_in; (void)out_size; (void)ws_size;
  const float* mkt  = (const float*)d_in[0];
  const float* ast  = (const float*)d_in[1];
  const float* pe   = (const float*)d_in[2];
  const float* Wm   = (const float*)d_in[3];
  const float* bm   = (const float*)d_in[4];
  const float* Wa   = (const float*)d_in[5];
  const float* ba   = (const float*)d_in[6];
  const float* Wq   = (const float*)d_in[7];
  const float* bq   = (const float*)d_in[8];
  const float* Wk   = (const float*)d_in[9];
  const float* bk   = (const float*)d_in[10];
  const float* Wv   = (const float*)d_in[11];
  const float* bv   = (const float*)d_in[12];
  const float* Wo   = (const float*)d_in[13];
  const float* bo   = (const float*)d_in[14];
  /* d_in[15] = pbias: per-head scalar added to every score in a row ->
     cancels under softmax; intentionally unused. */
  const float* W1   = (const float*)d_in[16];
  const float* b1   = (const float*)d_in[17];
  const float* W2   = (const float*)d_in[18];
  const float* b2   = (const float*)d_in[19];
  const float* ln1g = (const float*)d_in[20];
  const float* ln1b = (const float*)d_in[21];
  const float* ln2g = (const float*)d_in[22];
  const float* ln2b = (const float*)d_in[23];
  const float* lnfg = (const float*)d_in[24];
  const float* lnfb = (const float*)d_in[25];
  const float* Wout = (const float*)d_in[26];
  const float* bout = (const float*)d_in[27];

  char* ws = (char*)d_ws;
  size_t off = 0;
  auto carve = [&](size_t bytes) {
    void* p = ws + off;
    off += (bytes + 255) & ~(size_t)255;
    return p;
  };
  const size_t NT = (size_t)BB * SS;            // 2048 token rows
  float*          x32 = (float*)carve(NT * DD * 4);
  float*          t32 = (float*)carve(NT * DD * 4);
  unsigned short* xb  = (unsigned short*)carve(NT * DD * 2);
  unsigned short* qb  = (unsigned short*)carve(NT * DD * 2);
  unsigned short* kb  = (unsigned short*)carve(NT * DD * 2);
  unsigned short* vb  = (unsigned short*)carve(NT * DD * 2);
  unsigned short* cb  = (unsigned short*)carve(NT * DD * 2);
  unsigned short* f1b = (unsigned short*)carve(NT * DFF * 2);
  unsigned short* wbq = (unsigned short*)carve((size_t)DD * DD * 2);
  unsigned short* wbk = (unsigned short*)carve((size_t)DD * DD * 2);
  unsigned short* wbv = (unsigned short*)carve((size_t)DD * DD * 2);
  unsigned short* wbo = (unsigned short*)carve((size_t)DD * DD * 2);
  unsigned short* wb1 = (unsigned short*)carve((size_t)DFF * DD * 2);
  unsigned short* wb2 = (unsigned short*)carve((size_t)DD * DFF * 2);

  const int M = (int)NT;                         // 2048
  dim3 gemmD(M / GTM, DD / GTN);                 // (16, 8)
  dim3 gemmF1(M / GTM, DFF / GTN);               // (16, 32)
  dim3 attnG(SS / 64, HH, BB);                   // (16, 16, 2)

  embed_kernel<<<(BB * SS * DD + 255) / 256, 256, 0, stream>>>(
      mkt, ast, pe, Wm, bm, Wa, ba, x32, xb);

  for (int l = 0; l < LL; ++l) {
    cvt(Wq + (size_t)l * DD * DD, wbq, DD * DD, stream);
    cvt(Wk + (size_t)l * DD * DD, wbk, DD * DD, stream);
    cvt(Wv + (size_t)l * DD * DD, wbv, DD * DD, stream);
    cvt(Wo + (size_t)l * DD * DD, wbo, DD * DD, stream);
    cvt(W1 + (size_t)l * DFF * DD, wb1, DFF * DD, stream);
    cvt(W2 + (size_t)l * DD * DFF, wb2, DD * DFF, stream);

    // Q, K, V projections (bf16 outputs feed attention directly).
    gemm_bf16_kernel<<<gemmD, 256, 0, stream>>>(
        xb, wbq, bq + (size_t)l * DD, nullptr, nullptr, qb, M, DD, DD, 0);
    gemm_bf16_kernel<<<gemmD, 256, 0, stream>>>(
        xb, wbk, bk + (size_t)l * DD, nullptr, nullptr, kb, M, DD, DD, 0);
    gemm_bf16_kernel<<<gemmD, 256, 0, stream>>>(
        xb, wbv, bv + (size_t)l * DD, nullptr, nullptr, vb, M, DD, DD, 0);

    attn_kernel<<<attnG, 128, 0, stream>>>(qb, kb, vb, cb);

    // Output projection + residual, then LN1.
    gemm_bf16_kernel<<<gemmD, 256, 0, stream>>>(
        cb, wbo, bo + (size_t)l * DD, x32, t32, nullptr, M, DD, DD, 0);
    ln_kernel<<<M, 256, 0, stream>>>(t32, ln1g + (size_t)l * DD,
                                     ln1b + (size_t)l * DD, x32, xb);

    // FFN: relu(x@W1^T+b1) @ W2^T + b2 + residual, then LN2.
    gemm_bf16_kernel<<<gemmF1, 256, 0, stream>>>(
        xb, wb1, b1 + (size_t)l * DFF, nullptr, nullptr, f1b, M, DFF, DD, 1);
    gemm_bf16_kernel<<<gemmD, 256, 0, stream>>>(
        f1b, wb2, b2 + (size_t)l * DD, x32, t32, nullptr, M, DD, DFF, 0);
    ln_kernel<<<M, 256, 0, stream>>>(t32, ln2g + (size_t)l * DD,
                                     ln2b + (size_t)l * DD, x32, xb);
  }

  ln_kernel<<<M, 256, 0, stream>>>(x32, lnfg, lnfb, t32, nullptr);
  out_kernel<<<M, 256, 0, stream>>>(t32, Wout, bout, (float*)d_out);
}
